// DifferentialAttention_72559177498894
// MI455X (gfx1250) — matrix operand order repaired and checked by reading, not compiled, not run on hardware
//
#include <hip/hip_runtime.h>

// ---------------- problem constants ----------------
#define B_ 2
#define S_ 2048
#define HS_ 1024
#define H_ 16
#define KV_ 4
#define D_ 64
#define GROUPS_ (H_ / KV_)
#define ROWS_ (B_ * S_)          // 4096 rows of the token dimension
#define NQ_ (2 * HS_)            // 2048
#define NK_ (2 * KV_ * D_)       // 512
#define NV_ (KV_ * D_)           // 256

typedef __attribute__((ext_vector_type(16))) __bf16 v16bf;
typedef __attribute__((ext_vector_type(8)))  float  v8f;

union Frag {
  v16bf v;
  uint4 q[2];
  unsigned short us[16];
};

__device__ __forceinline__ unsigned short f2bf(float f) {
  unsigned int u = __float_as_uint(f);
  u += 0x7FFFu + ((u >> 16) & 1u);   // round-to-nearest-even
  return (unsigned short)(u >> 16);
}
__device__ __forceinline__ float bf2f(unsigned short h) {
  return __uint_as_float(((unsigned int)h) << 16);
}

// A-fragment: 16x32 bf16, row-major source, lda in elements.
// lane<16: M=lane, K = {0..7, 16..23}; lane>=16: M=lane-16, K = {8..15, 24..31}
__device__ __forceinline__ void load_a_frag(Frag& f, const unsigned short* A,
                                            int lda, int row0, int k0, int lane) {
  int m = lane & 15, hf = lane >> 4;
  const unsigned short* p = A + (size_t)(row0 + m) * lda + k0 + hf * 8;
  f.q[0] = *(const uint4*)p;
  f.q[1] = *(const uint4*)(p + 16);
}
// B-fragment: 32x16 bf16 from transposed storage Bt[N][K] (ldb in elements).
// lane<16: N=lane, K = {0..7, 16..23}; lane>=16: N=lane-16, K = {8..15, 24..31} (2 bf16 per VGPR)
__device__ __forceinline__ void load_b_frag(Frag& f, const unsigned short* Bt,
                                            int ldb, int col0, int k0, int lane) {
  int n = lane & 15, koff = (lane >> 4) << 3;
  const unsigned short* p = Bt + (size_t)(col0 + n) * ldb + k0 + koff;
  f.q[0] = *(const uint4*)p;
  f.q[1] = *(const uint4*)(p + 16);
}
__device__ __forceinline__ v8f wmma_bf16(const Frag& a, const Frag& b, v8f c) {
  return __builtin_amdgcn_wmma_f32_16x16x32_bf16(false, a.v, false, b.v,
                                                 (short)0, c, false, false);
}

// Async DMA of one 16-byte chunk global -> LDS (ASYNCcnt-tracked).
// VDST = VGPR holding LDS byte address; VADDR = 64-bit global address.
__device__ __forceinline__ void async_g2l_b128(unsigned lds_addr, const void* gptr) {
  asm volatile("global_load_async_to_lds_b128 %0, %1, off"
               :: "v"(lds_addr), "v"((unsigned long long)(size_t)gptr)
               : "memory");
}
__device__ __forceinline__ void wait_asynccnt0() {
  asm volatile("s_wait_asynccnt 0x0" ::: "memory");
}

// ---------------- conversion kernels ----------------
__global__ void cvt_f32_bf16_kernel(const float* __restrict__ x,
                                    unsigned short* __restrict__ y, size_t n) {
  size_t i = (size_t)blockIdx.x * blockDim.x + threadIdx.x;
  if (i < n) y[i] = f2bf(x[i]);
}

// Wt[n*K + k] = bf16(W[k*N + n])
__global__ void transpose_cvt_kernel(const float* __restrict__ W,
                                     unsigned short* __restrict__ Wt,
                                     int K, int N) {
  size_t i = (size_t)blockIdx.x * blockDim.x + threadIdx.x;
  size_t total = (size_t)K * N;
  if (i >= total) return;
  int n = (int)(i / K), k = (int)(i % K);
  Wt[i] = f2bf(W[(size_t)k * N + n]);
}

__global__ void lam_kernel(const float* lq1, const float* lk1,
                           const float* lq2, const float* lk2, float* lam) {
  __shared__ float s1[D_], s2[D_];
  int t = threadIdx.x;
  s1[t] = lq1[t] * lk1[t];
  s2[t] = lq2[t] * lk2[t];
  __syncthreads();
  if (t == 0) {
    float a = 0.f, b = 0.f;
    for (int i = 0; i < D_; i++) { a += s1[i]; b += s2[i]; }
    float lambda_init = 0.8f - 0.6f * __expf(-0.3f);
    *lam = __expf(a) - __expf(b) + lambda_init;
  }
}

// ---------------- generic WMMA GEMM: C[M,N] = A[M,K](bf16) x Bt[N,K](bf16) ----------------
__global__ __launch_bounds__(32) void wmma_gemm_kernel(
    const unsigned short* __restrict__ A, const unsigned short* __restrict__ Bt,
    void* __restrict__ Cout, int M, int N, int K, int f32out) {
  int lane = threadIdx.x;
  int nBase = blockIdx.x * 64;
  int mBase = blockIdx.y * 16;
  v8f z = {0.f, 0.f, 0.f, 0.f, 0.f, 0.f, 0.f, 0.f};
  v8f acc[4];
#pragma unroll
  for (int t = 0; t < 4; t++) acc[t] = z;

  for (int k0 = 0; k0 < K; k0 += 32) {
    Frag a;
    load_a_frag(a, A, K, mBase, k0, lane);
    if (k0 + 32 < K)  // keep the A stream hot: global_prefetch_b8 (near-cache fill)
      __builtin_prefetch((const void*)(A + (size_t)(mBase + (lane & 15)) * K + k0 + 32), 0, 3);
#pragma unroll
    for (int t = 0; t < 4; t++) {
      Frag b;
      load_b_frag(b, Bt, K, nBase + 16 * t, k0, lane);
      acc[t] = wmma_bf16(a, b, acc[t]);
    }
  }
  int colL = lane & 15, rb = (lane >> 4) * 8;
#pragma unroll
  for (int t = 0; t < 4; t++)
#pragma unroll
    for (int r = 0; r < 8; r++) {
      size_t idx = (size_t)(mBase + rb + r) * N + nBase + 16 * t + colL;
      if (f32out) ((float*)Cout)[idx] = acc[t][r];
      else        ((unsigned short*)Cout)[idx] = f2bf(acc[t][r]);
    }
}

// ---------------- RoPE / layout kernels ----------------
// qraw[B*S, 2H*D] -> Q[(st*B+b)*H+h][S][D] with RoPE
__global__ void rope_q_kernel(const unsigned short* __restrict__ qraw,
                              const float* __restrict__ cosT, const float* __restrict__ sinT,
                              unsigned short* __restrict__ Q) {
  size_t tid = (size_t)blockIdx.x * blockDim.x + threadIdx.x;
  const size_t total = (size_t)2 * B_ * H_ * S_ * D_;
  if (tid >= total) return;
  int d = (int)(tid % D_); size_t t = tid / D_;
  int s = (int)(t % S_); t /= S_;
  int h = (int)(t % H_); t /= H_;
  int b = (int)(t % B_); int st = (int)(t / B_);
  int head = st * H_ + h;
  size_t rbase = ((size_t)(b * S_ + s)) * NQ_ + (size_t)head * D_;
  float v0 = bf2f(qraw[rbase + d]);
  float out;
  if (d < 32) {
    float v2 = bf2f(qraw[rbase + d + 32]);
    out = v0 * cosT[s * 32 + d] - v2 * sinT[s * 32 + d];
  } else {
    float v1 = bf2f(qraw[rbase + d - 32]);
    out = v0 * cosT[s * 32 + d - 32] + v1 * sinT[s * 32 + d - 32];
  }
  Q[tid] = f2bf(out);
}

// kraw[B*S, 2KV*D] -> K[(st*B+b)*KV+kv][S][D] with RoPE
__global__ void rope_k_kernel(const unsigned short* __restrict__ kraw,
                              const float* __restrict__ cosT, const float* __restrict__ sinT,
                              unsigned short* __restrict__ Kd) {
  size_t tid = (size_t)blockIdx.x * blockDim.x + threadIdx.x;
  const size_t total = (size_t)2 * B_ * KV_ * S_ * D_;
  if (tid >= total) return;
  int d = (int)(tid % D_); size_t t = tid / D_;
  int s = (int)(t % S_); t /= S_;
  int kv = (int)(t % KV_); t /= KV_;
  int b = (int)(t % B_); int st = (int)(t / B_);
  int head = st * KV_ + kv;
  size_t rbase = ((size_t)(b * S_ + s)) * NK_ + (size_t)head * D_;
  float v0 = bf2f(kraw[rbase + d]);
  float out;
  if (d < 32) {
    float v2 = bf2f(kraw[rbase + d + 32]);
    out = v0 * cosT[s * 32 + d] - v2 * sinT[s * 32 + d];
  } else {
    float v1 = bf2f(kraw[rbase + d - 32]);
    out = v0 * cosT[s * 32 + d - 32] + v1 * sinT[s * 32 + d - 32];
  }
  Kd[tid] = f2bf(out);
}

// vraw[B*S, KV*D] -> Vt[(b*KV+kv)*D + d][S]  (transposed for contiguous PV B-fragments)
__global__ void vtrans_kernel(const unsigned short* __restrict__ vraw,
                              unsigned short* __restrict__ Vt) {
  size_t tid = (size_t)blockIdx.x * blockDim.x + threadIdx.x;
  const size_t total = (size_t)B_ * KV_ * D_ * S_;
  if (tid >= total) return;
  int s = (int)(tid % S_); size_t t = tid / S_;
  int d = (int)(t % D_); t /= D_;
  int kv = (int)(t % KV_); int b = (int)(t / KV_);
  Vt[tid] = vraw[((size_t)(b * S_ + s)) * NV_ + kv * D_ + d];
}

// ---------------- flash differential attention ----------------
// grid (S/16, B*H), 1 wave per 16-query tile of one head.
// Per 32-key block: V tile (64d x 32k bf16, 4KB) is DMA'd to LDS with
// global_load_async_to_lds_b128 (overlapped with the QK^T WMMAs + softmax),
// then the 8 PV WMMAs read their B-fragments from LDS.
__global__ __launch_bounds__(32) void diff_attn_kernel(
    const unsigned short* __restrict__ Qr, const unsigned short* __restrict__ Kr,
    const unsigned short* __restrict__ Vt, const float* __restrict__ lamPtr,
    const float* __restrict__ subln, unsigned short* __restrict__ attnOut) {
  __shared__ alignas(16) unsigned short Plds[2][16 * 32];
  __shared__ alignas(16) unsigned short Vtile[D_ * 32];  // [d][k] for one key block

  int lane = threadIdx.x;
  int qt = blockIdx.x;
  int bh = blockIdx.y;
  int b = bh / H_, h = bh % H_;
  int kv = h / GROUPS_;
  int qbase = qt * 16;
  int hf = lane >> 4, colL = lane & 15;

  const unsigned short* Qs[2];
  const unsigned short* Ks[2];
  Qs[0] = Qr + ((size_t)((0 * B_ + b) * H_ + h)) * S_ * D_;
  Qs[1] = Qr + ((size_t)((1 * B_ + b) * H_ + h)) * S_ * D_;
  Ks[0] = Kr + ((size_t)((0 * B_ + b) * KV_ + kv)) * S_ * D_;
  Ks[1] = Kr + ((size_t)((1 * B_ + b) * KV_ + kv)) * S_ * D_;
  const unsigned short* Vh = Vt + ((size_t)(b * KV_ + kv)) * D_ * S_;

  Frag qa[2][2];
#pragma unroll
  for (int s = 0; s < 2; s++) {
    load_a_frag(qa[s][0], Qs[s], D_, qbase, 0, lane);
    load_a_frag(qa[s][1], Qs[s], D_, qbase, 32, lane);
  }

  v8f z = {0.f, 0.f, 0.f, 0.f, 0.f, 0.f, 0.f, 0.f};
  v8f O[2][4];
  float m[2][8], l[2][8];
#pragma unroll
  for (int s = 0; s < 2; s++) {
#pragma unroll
    for (int t = 0; t < 4; t++) O[s][t] = z;
#pragma unroll
    for (int r = 0; r < 8; r++) { m[s][r] = -1e30f; l[s][r] = 0.f; }
  }

  // per-lane assignment for the async V-tile copy: 8 ops x (8 rows x 32 cols)
  int vrow0 = lane >> 2;            // 0..7
  int vcolp = (lane & 3) * 8;       // 0,8,16,24 (x 8 bf16 = 16B chunks)

  int nkb = qbase / 32 + 1;  // causal: keys <= qbase+15
  for (int kb = 0; kb < nkb; ++kb) {
    int kbase = kb * 32;

    // kick off async DMA of the V tile into LDS (tracked by ASYNCcnt)
#pragma unroll
    for (int i = 0; i < 8; i++) {
      int drow = i * 8 + vrow0;
      unsigned laddr = (unsigned)(size_t)&Vtile[drow * 32 + vcolp];
      async_g2l_b128(laddr, (const void*)(Vh + (size_t)drow * S_ + kbase + vcolp));
    }

#pragma unroll
    for (int s = 0; s < 2; s++) {
      Frag b00, b01, b10, b11;
      load_b_frag(b00, Ks[s], D_, kbase,      0,  lane);
      load_b_frag(b01, Ks[s], D_, kbase,      32, lane);
      load_b_frag(b10, Ks[s], D_, kbase + 16, 0,  lane);
      load_b_frag(b11, Ks[s], D_, kbase + 16, 32, lane);
      v8f clo = wmma_bf16(qa[s][0], b00, z);
      clo = wmma_bf16(qa[s][1], b01, clo);
      v8f chi = wmma_bf16(qa[s][0], b10, z);
      chi = wmma_bf16(qa[s][1], b11, chi);
#pragma unroll
      for (int r = 0; r < 8; r++) {
        int rowg = qbase + hf * 8 + r;
        float vlo = clo[r] * 0.125f;
        if (kbase + colL > rowg) vlo = -1e30f;
        float vhi = chi[r] * 0.125f;
        if (kbase + 16 + colL > rowg) vhi = -1e30f;
        float tmax = fmaxf(vlo, vhi);
#pragma unroll
        for (int off = 1; off < 16; off <<= 1)
          tmax = fmaxf(tmax, __shfl_xor(tmax, off, 32));
        float mo = m[s][r];
        float mn = fmaxf(mo, tmax);
        float alpha = __expf(mo - mn);
        m[s][r] = mn;
        float plo = __expf(vlo - mn);
        float phi = __expf(vhi - mn);
        float rs = plo + phi;
#pragma unroll
        for (int off = 1; off < 16; off <<= 1)
          rs += __shfl_xor(rs, off, 32);
        l[s][r] = l[s][r] * alpha + rs;
#pragma unroll
        for (int t = 0; t < 4; t++) O[s][t][r] *= alpha;
        Plds[s][(hf * 8 + r) * 32 + colL] = f2bf(plo);
        Plds[s][(hf * 8 + r) * 32 + 16 + colL] = f2bf(phi);
      }
    }
    __syncthreads();
    wait_asynccnt0();  // V tile is now resident in LDS
    Frag vb[4];
#pragma unroll
    for (int t = 0; t < 4; t++)
      load_b_frag(vb[t], &Vtile[0], 32, 16 * t, 0, lane);
#pragma unroll
    for (int s = 0; s < 2; s++) {
      Frag pa;
      load_a_frag(pa, &Plds[s][0], 32, 0, 0, lane);
#pragma unroll
      for (int t = 0; t < 4; t++) O[s][t] = wmma_bf16(pa, vb[t], O[s][t]);
    }
    __syncthreads();
  }

  float lam = lamPtr[0];
  float o[4][8];
#pragma unroll
  for (int t = 0; t < 4; t++)
#pragma unroll
    for (int r = 0; r < 8; r++)
      o[t][r] = O[0][t][r] / l[0][r] - lam * O[1][t][r] / l[1][r];

#pragma unroll
  for (int r = 0; r < 8; r++) {
    float ss = 0.f;
#pragma unroll
    for (int t = 0; t < 4; t++) ss += o[t][r] * o[t][r];
#pragma unroll
    for (int off = 1; off < 16; off <<= 1)
      ss += __shfl_xor(ss, off, 32);
    float inv = rsqrtf(ss * (1.0f / 64.0f) + 1e-5f);
    int rowg = qbase + hf * 8 + r;
#pragma unroll
    for (int t = 0; t < 4; t++) {
      float w = subln[16 * t + colL];
      attnOut[(size_t)(b * S_ + rowg) * HS_ + h * D_ + 16 * t + colL] =
          f2bf(o[t][r] * inv * w);
    }
  }
}

// ---------------- host launch ----------------
extern "C" void kernel_launch(void* const* d_in, const int* in_sizes, int n_in,
                              void* d_out, int out_size, void* d_ws, size_t ws_size,
                              hipStream_t stream) {
  (void)in_sizes; (void)n_in; (void)out_size; (void)ws_size;
  const float* x        = (const float*)d_in[0];
  const float* rope_cos = (const float*)d_in[1];
  const float* rope_sin = (const float*)d_in[2];
  // d_in[3] attention_mask: causal mask applied analytically in-kernel
  const float* Wq   = (const float*)d_in[4];
  const float* Wk   = (const float*)d_in[5];
  const float* Wv   = (const float*)d_in[6];
  const float* Wo   = (const float*)d_in[7];
  const float* lq1  = (const float*)d_in[8];
  const float* lk1  = (const float*)d_in[9];
  const float* lq2  = (const float*)d_in[10];
  const float* lk2  = (const float*)d_in[11];
  const float* subln = (const float*)d_in[12];
  float* out = (float*)d_out;

  char* ws = (char*)d_ws;
  size_t off = 0;
  auto take = [&](size_t bytes) -> char* {
    char* p = ws + off;
    off = (off + bytes + 255) & ~(size_t)255;
    return p;
  };
  unsigned short* xbf  = (unsigned short*)take((size_t)ROWS_ * HS_ * 2);
  unsigned short* WqT  = (unsigned short*)take((size_t)NQ_ * HS_ * 2);
  unsigned short* WkT  = (unsigned short*)take((size_t)NK_ * HS_ * 2);
  unsigned short* WvT  = (unsigned short*)take((size_t)NV_ * HS_ * 2);
  unsigned short* WoT  = (unsigned short*)take((size_t)HS_ * HS_ * 2);
  unsigned short* qraw = (unsigned short*)take((size_t)ROWS_ * NQ_ * 2);
  unsigned short* kraw = (unsigned short*)take((size_t)ROWS_ * NK_ * 2);
  unsigned short* vraw = (unsigned short*)take((size_t)ROWS_ * NV_ * 2);
  unsigned short* Qr   = (unsigned short*)take((size_t)2 * B_ * H_ * S_ * D_ * 2);
  unsigned short* Kr   = (unsigned short*)take((size_t)2 * B_ * KV_ * S_ * D_ * 2);
  unsigned short* Vtb  = (unsigned short*)take((size_t)B_ * KV_ * D_ * S_ * 2);
  unsigned short* attnO = (unsigned short*)take((size_t)ROWS_ * HS_ * 2);
  float* lamW = (float*)take(sizeof(float));

  const int T = 256;
  // 1) precision conversion / weight transposes (one-time, L2-resident)
  {
    size_t n = (size_t)ROWS_ * HS_;
    cvt_f32_bf16_kernel<<<dim3((unsigned)((n + T - 1) / T)), T, 0, stream>>>(x, xbf, n);
  }
  transpose_cvt_kernel<<<dim3((unsigned)(((size_t)HS_ * NQ_ + T - 1) / T)), T, 0, stream>>>(Wq, WqT, HS_, NQ_);
  transpose_cvt_kernel<<<dim3((unsigned)(((size_t)HS_ * NK_ + T - 1) / T)), T, 0, stream>>>(Wk, WkT, HS_, NK_);
  transpose_cvt_kernel<<<dim3((unsigned)(((size_t)HS_ * NV_ + T - 1) / T)), T, 0, stream>>>(Wv, WvT, HS_, NV_);
  transpose_cvt_kernel<<<dim3((unsigned)(((size_t)HS_ * HS_ + T - 1) / T)), T, 0, stream>>>(Wo, WoT, HS_, HS_);
  lam_kernel<<<1, D_, 0, stream>>>(lq1, lk1, lq2, lk2, lamW);

  // 2) QKV projections (WMMA)
  wmma_gemm_kernel<<<dim3(NQ_ / 64, ROWS_ / 16), 32, 0, stream>>>(xbf, WqT, qraw, ROWS_, NQ_, HS_, 0);
  wmma_gemm_kernel<<<dim3(NK_ / 64, ROWS_ / 16), 32, 0, stream>>>(xbf, WkT, kraw, ROWS_, NK_, HS_, 0);
  wmma_gemm_kernel<<<dim3(NV_ / 64, ROWS_ / 16), 32, 0, stream>>>(xbf, WvT, vraw, ROWS_, NV_, HS_, 0);

  // 3) RoPE + attention layouts
  {
    size_t n = (size_t)2 * B_ * H_ * S_ * D_;
    rope_q_kernel<<<dim3((unsigned)((n + T - 1) / T)), T, 0, stream>>>(qraw, rope_cos, rope_sin, Qr);
  }
  {
    size_t n = (size_t)2 * B_ * KV_ * S_ * D_;
    rope_k_kernel<<<dim3((unsigned)((n + T - 1) / T)), T, 0, stream>>>(kraw, rope_cos, rope_sin, Kr);
  }
  {
    size_t n = (size_t)B_ * KV_ * D_ * S_;
    vtrans_kernel<<<dim3((unsigned)((n + T - 1) / T)), T, 0, stream>>>(vraw, Vtb);
  }

  // 4) flash differential attention (WMMA QK^T + async-LDS V staging + WMMA PV)
  diff_attn_kernel<<<dim3(S_ / 16, B_ * H_), 32, 0, stream>>>(Qr, Kr, Vtb, lamW, subln, attnO);

  // 5) output projection (WMMA, fp32 out)
  wmma_gemm_kernel<<<dim3(HS_ / 64, ROWS_ / 16), 32, 0, stream>>>(attnO, WoT, out, ROWS_, HS_, HS_, 1);
}
